// MultiHeadAttention_75763223101899
// MI455X (gfx1250) — matrix operand order repaired and checked by reading, not compiled, not run on hardware
//
#include <hip/hip_runtime.h>
#include <hip/hip_bf16.h>

// ---------------------------------------------------------------------------
// MHA block for MI455X (gfx1250): all GEMMs via v_wmma_f32_16x16x32_f16.
// HBM-bound on the 512MB attention-probability output (write-once -> NT
// stores so it doesn't evict reused data from the 192MB L2). Weights are
// converted to f16 once; GEMM inner loops are pure f16 b128 loads + WMMA.
// ---------------------------------------------------------------------------

typedef __attribute__((ext_vector_type(16))) _Float16 v16h;
typedef __attribute__((ext_vector_type(8)))  _Float16 v8h;
typedef __attribute__((ext_vector_type(4)))  _Float16 v4h;
typedef __attribute__((ext_vector_type(8)))  float    v8f;
typedef __attribute__((ext_vector_type(4)))  float    f4;
typedef __attribute__((ext_vector_type(4)))  int      i4;

#define B_  2
#define S_  2048
#define D_  1024
#define H_  16
#define HD_ 64

static __device__ inline v8f wmma32(v16h a, v16h b, v8f c) {
  return __builtin_amdgcn_wmma_f32_16x16x32_f16(false, a, false, b, (short)0, c,
                                                false, false);
}

static __device__ inline v16h frag_from_2x8(v8h lo, v8h hi8) {
  v16h r;
#pragma unroll
  for (int j = 0; j < 8; ++j) { r[j] = lo[j]; r[j + 8] = hi8[j]; }
  return r;
}
// A fragment (16x32 f16): halves 0..7 <- K = 8*hi+0..7, 8..15 <- K = 16+8*hi+0..7
static __device__ inline v16h frag_a_f16(const _Float16* __restrict__ p, int hi) {
  return frag_from_2x8(*(const v8h*)(p + 8 * hi), *(const v8h*)(p + 16 + 8 * hi));
}
// B fragment (32x16 f16): halves 0..7 <- K = 8*hi+0..7, 8..15 <- K = 16+8*hi+0..7 (source row = N of stored W)
static __device__ inline v16h frag_b_f16(const _Float16* __restrict__ p, int hi) {
  const _Float16* q = p + 8 * hi;
  return frag_from_2x8(*(const v8h*)q, *(const v8h*)(q + 16));
}
static __device__ inline v8h cvt8(f4 a, f4 b) {
  v8h r;
#pragma unroll
  for (int j = 0; j < 4; ++j) { r[j] = (_Float16)a[j]; r[j + 4] = (_Float16)b[j]; }
  return r;
}
static __device__ inline v16h frag_a_f32(const float* __restrict__ p, int hi) {
  const float* p0 = p + 8 * hi;
  const float* p1 = p + 16 + 8 * hi;
  f4 a0 = *(const f4*)p0, a1 = *(const f4*)(p0 + 4);
  f4 b0 = *(const f4*)p1, b1 = *(const f4*)(p1 + 4);
  return frag_from_2x8(cvt8(a0, a1), cvt8(b0, b1));
}

// ---------------------------------------------------------------------------
// Kernel 0: f32 -> f16 conversion (weights; runs once, tiny)
// ---------------------------------------------------------------------------
__global__ void cvt_f32_f16(const float* __restrict__ src, _Float16* __restrict__ dst,
                            int n4) {
  const int i = blockIdx.x * blockDim.x + threadIdx.x;
  if (i < n4) {
    const f4 v = ((const f4*)src)[i];
    v4h o;
#pragma unroll
    for (int j = 0; j < 4; ++j) o[j] = (_Float16)v[j];
    ((v4h*)dst)[i] = o;
  }
}

// ---------------------------------------------------------------------------
// Kernel 1: Y = X @ W.T + bias  (X f32 [4096,1024], Wh f16 [1024,1024])
// transposed==0 -> [B,H,S,HD] f16; transposed==1 -> [B,H,HD,S] f16 (for V).
// grid (N/64, M/32), block 64 (2 waves, 16 rows each).
// ---------------------------------------------------------------------------
__global__ void proj_gemm(const float* __restrict__ X, const _Float16* __restrict__ Wh,
                          const float* __restrict__ bias, _Float16* __restrict__ out,
                          int transposed, float scale) {
  const int lane = threadIdx.x & 31, wv = threadIdx.x >> 5;
  const int laneM = lane & 15, hi = lane >> 4;
  const int n0 = blockIdx.x * 64;
  const int m0 = (blockIdx.y * 2 + wv) * 16;

  v8f acc[4] = {};
  const float* arow = X + (size_t)(m0 + laneM) * D_;
  const _Float16* brow0 = Wh + (size_t)(n0 + laneM) * D_;
  const _Float16* brow1 = brow0 + (size_t)16 * D_;
  const _Float16* brow2 = brow0 + (size_t)32 * D_;
  const _Float16* brow3 = brow0 + (size_t)48 * D_;

  for (int k0 = 0; k0 < D_; k0 += 32) {
    v16h bf0 = frag_b_f16(brow0 + k0, hi);
    v16h bf1 = frag_b_f16(brow1 + k0, hi);
    v16h bf2 = frag_b_f16(brow2 + k0, hi);
    v16h bf3 = frag_b_f16(brow3 + k0, hi);
    v16h a   = frag_a_f32(arow + k0, hi);
    acc[0] = wmma32(a, bf0, acc[0]);
    acc[1] = wmma32(a, bf1, acc[1]);
    acc[2] = wmma32(a, bf2, acc[2]);
    acc[3] = wmma32(a, bf3, acc[3]);
  }
#pragma unroll
  for (int nt = 0; nt < 4; ++nt) {
    const int n = n0 + nt * 16 + laneM;
    const float bv = bias[n];
    const int h = n / HD_, dd = n % HD_;
    if (!transposed) {
#pragma unroll
      for (int r = 0; r < 8; ++r) {
        const int m = m0 + r + 8 * hi;
        const int b = m / S_, s = m % S_;
        out[((size_t)(b * H_ + h) * S_ + s) * HD_ + dd] =
            (_Float16)((acc[nt][r] + bv) * scale);
      }
    } else {
      // rows are contiguous along s in [B,H,HD,S]: pack 8 halves, one b128 store
      const int m = m0 + 8 * hi;           // first row of this lane's 8
      const int b = m / S_, s = m % S_;
      v8h pk;
#pragma unroll
      for (int r = 0; r < 8; ++r) pk[r] = (_Float16)((acc[nt][r] + bv) * scale);
      *(v8h*)(out + ((size_t)(b * H_ + h) * HD_ + dd) * S_ + s) = pk;
    }
  }
}

// ---------------------------------------------------------------------------
// Kernel 2: fused scores -> mask -> softmax -> attention write -> context.
// grid (S/32, B*H), block 64 (2 waves); 16x2048 f32 strip per wave in LDS.
// ---------------------------------------------------------------------------
__global__ void attn_kernel(const _Float16* __restrict__ qh,
                            const _Float16* __restrict__ kh,
                            const _Float16* __restrict__ vT,
                            const int* __restrict__ mask,
                            float* __restrict__ attOut,
                            _Float16* __restrict__ ctx) {
  extern __shared__ float lds[];
  const int lane = threadIdx.x & 31, wv = threadIdx.x >> 5;
  const int laneM = lane & 15, hi = lane >> 4;
  const int bh = blockIdx.y;
  const int b = bh >> 4, h = bh & 15;
  const int row0 = blockIdx.x * 32 + wv * 16;

  float* P = lds + (size_t)wv * 16 * S_;

  const _Float16* qbase = qh + ((size_t)(b * H_ + h) * S_ + row0) * HD_;
  const _Float16* kbase = kh + (size_t)(b * H_ + h) * S_ * HD_;
  const _Float16* vbase = vT + (size_t)(b * H_ + h) * HD_ * S_;

  // ---- scores: Q (16x64) x K^T (64x2048); two key tiles per iteration ----
  const _Float16* arow = qbase + (size_t)laneM * HD_;
  const v16h a0 = frag_a_f16(arow + 0, hi);
  const v16h a1 = frag_a_f16(arow + 32, hi);
  for (int kt = 0; kt < S_; kt += 32) {
    const _Float16* browA = kbase + (size_t)(kt + laneM) * HD_;
    const _Float16* browB = browA + (size_t)16 * HD_;
    v16h bA0 = frag_b_f16(browA, hi);
    v16h bA1 = frag_b_f16(browA + 32, hi);
    v16h bB0 = frag_b_f16(browB, hi);
    v16h bB1 = frag_b_f16(browB + 32, hi);
    v8f cA = {}, cB = {};
    cA = wmma32(a0, bA0, cA);
    cA = wmma32(a1, bA1, cA);
    cB = wmma32(a0, bB0, cB);
    cB = wmma32(a1, bB1, cB);
#pragma unroll
    for (int r = 0; r < 8; ++r) {
      P[(size_t)(r + 8 * hi) * S_ + kt + laneM]      = cA[r];
      P[(size_t)(r + 8 * hi) * S_ + kt + 16 + laneM] = cB[r];
    }
  }

  // ---- mask + softmax per row, float4-vectorized; write attention f32 ----
  for (int r = 0; r < 16; ++r) {
    const int qrow = row0 + r;
    f4* prow = (f4*)(P + (size_t)r * S_);
    const i4* mr = (const i4*)(mask + ((size_t)b * S_ + qrow) * S_);
    f4 mxv = {-3.0e38f, -3.0e38f, -3.0e38f, -3.0e38f};
    for (int t = lane; t < S_ / 4; t += 32) {
      f4 sv = prow[t];
      const i4 mk = mr[t];
#pragma unroll
      for (int j = 0; j < 4; ++j) {
        if (mk[j] == 0) sv[j] = -1.0e9f;
        mxv[j] = fmaxf(mxv[j], sv[j]);
      }
      prow[t] = sv;
    }
    float mx = fmaxf(fmaxf(mxv[0], mxv[1]), fmaxf(mxv[2], mxv[3]));
#pragma unroll
    for (int o = 16; o > 0; o >>= 1) mx = fmaxf(mx, __shfl_xor(mx, o, 32));
    f4 sumv = {0.f, 0.f, 0.f, 0.f};
    for (int t = lane; t < S_ / 4; t += 32) {
      f4 sv = prow[t];
      f4 e;
#pragma unroll
      for (int j = 0; j < 4; ++j) e[j] = __expf(sv[j] - mx);
      prow[t] = e;
      sumv += e;
    }
    float sum = (sumv[0] + sumv[1]) + (sumv[2] + sumv[3]);
#pragma unroll
    for (int o = 16; o > 0; o >>= 1) sum += __shfl_xor(sum, o, 32);
    const float inv = 1.0f / sum;
    f4* orow = (f4*)(attOut + ((size_t)(b * H_ + h) * S_ + qrow) * S_);
    for (int t = lane; t < S_ / 4; t += 32) {
      f4 p = prow[t];
#pragma unroll
      for (int j = 0; j < 4; ++j) p[j] *= inv;
      prow[t] = p;
      // dominant 512MB HBM stream, write-once -> non-temporal b128 store
      __builtin_nontemporal_store(p, &orow[t]);
    }
  }

  // ---- context: P (16x2048, LDS f32->f16) x V (2048x64) ----
  v8f acc[4] = {};
  const float* pr = P + (size_t)laneM * S_;
  const _Float16* vrow0 = vbase + (size_t)laneM * S_;
  for (int k0 = 0; k0 < S_; k0 += 32) {
    v16h bf0 = frag_b_f16(vrow0 + k0, hi);
    v16h bf1 = frag_b_f16(vrow0 + (size_t)16 * S_ + k0, hi);
    v16h bf2 = frag_b_f16(vrow0 + (size_t)32 * S_ + k0, hi);
    v16h bf3 = frag_b_f16(vrow0 + (size_t)48 * S_ + k0, hi);
    const float* pk = pr + k0;
    f4 x0 = *(const f4*)(pk + 8 * hi);
    f4 x1 = *(const f4*)(pk + 8 * hi + 4);
    f4 x2 = *(const f4*)(pk + 16 + 8 * hi);
    f4 x3 = *(const f4*)(pk + 16 + 8 * hi + 4);
    v16h a = frag_from_2x8(cvt8(x0, x1), cvt8(x2, x3));
    acc[0] = wmma32(a, bf0, acc[0]);
    acc[1] = wmma32(a, bf1, acc[1]);
    acc[2] = wmma32(a, bf2, acc[2]);
    acc[3] = wmma32(a, bf3, acc[3]);
  }
#pragma unroll
  for (int nt = 0; nt < 4; ++nt) {
    const int col = h * HD_ + nt * 16 + laneM;
#pragma unroll
    for (int r = 0; r < 8; ++r) {
      const int m = row0 + r + 8 * hi;
      ctx[((size_t)b * S_ + m) * D_ + col] = (_Float16)acc[nt][r];
    }
  }
}

// ---------------------------------------------------------------------------
// Kernel 3: x = ctx(f16) @ Wo.T(f16) + bo + residual -> f32 scratch
// ---------------------------------------------------------------------------
__global__ void oproj_gemm(const _Float16* __restrict__ A, const _Float16* __restrict__ Wh,
                           const float* __restrict__ bias, const float* __restrict__ resid,
                           float* __restrict__ out) {
  const int lane = threadIdx.x & 31, wv = threadIdx.x >> 5;
  const int laneM = lane & 15, hi = lane >> 4;
  const int n0 = blockIdx.x * 64;
  const int m0 = (blockIdx.y * 2 + wv) * 16;

  v8f acc[4] = {};
  const _Float16* arow = A + (size_t)(m0 + laneM) * D_;
  const _Float16* brow0 = Wh + (size_t)(n0 + laneM) * D_;
  for (int k0 = 0; k0 < D_; k0 += 32) {
    v16h bf0 = frag_b_f16(brow0 + k0, hi);
    v16h bf1 = frag_b_f16(brow0 + (size_t)16 * D_ + k0, hi);
    v16h bf2 = frag_b_f16(brow0 + (size_t)32 * D_ + k0, hi);
    v16h bf3 = frag_b_f16(brow0 + (size_t)48 * D_ + k0, hi);
    v16h a   = frag_a_f16(arow + k0, hi);
    acc[0] = wmma32(a, bf0, acc[0]);
    acc[1] = wmma32(a, bf1, acc[1]);
    acc[2] = wmma32(a, bf2, acc[2]);
    acc[3] = wmma32(a, bf3, acc[3]);
  }
#pragma unroll
  for (int nt = 0; nt < 4; ++nt) {
    const int n = n0 + nt * 16 + laneM;
    const float bv = bias[n];
#pragma unroll
    for (int r = 0; r < 8; ++r) {
      const int m = m0 + r + 8 * hi;
      out[(size_t)m * D_ + n] = acc[nt][r] + bv + resid[(size_t)m * D_ + n];
    }
  }
}

// ---------------------------------------------------------------------------
// Kernel 4: row-wise LayerNorm over D=1024; one float4 per thread.
// ---------------------------------------------------------------------------
__global__ void layernorm_kernel(const float* __restrict__ x,
                                 const float* __restrict__ gamma,
                                 const float* __restrict__ beta,
                                 float* __restrict__ out) {
  __shared__ float rs[8], rs2[8], mv[2];
  const int row = blockIdx.x;
  const int lane = threadIdx.x & 31, wv = threadIdx.x >> 5;
  const f4* xr = (const f4*)(x + (size_t)row * D_);
  const f4 v = xr[threadIdx.x];
  float s  = (v[0] + v[1]) + (v[2] + v[3]);
  float s2 = (v[0] * v[0] + v[1] * v[1]) + (v[2] * v[2] + v[3] * v[3]);
#pragma unroll
  for (int o = 16; o > 0; o >>= 1) {
    s += __shfl_xor(s, o, 32);
    s2 += __shfl_xor(s2, o, 32);
  }
  if (lane == 0) { rs[wv] = s; rs2[wv] = s2; }
  __syncthreads();
  if (threadIdx.x == 0) {
    float ts = 0.f, ts2 = 0.f;
#pragma unroll
    for (int i = 0; i < 8; ++i) { ts += rs[i]; ts2 += rs2[i]; }
    const float mu = ts * (1.0f / D_);
    mv[0] = mu;
    mv[1] = ts2 * (1.0f / D_) - mu * mu;
  }
  __syncthreads();
  const float mu = mv[0];
  const float inv = rsqrtf(mv[1] + 1e-5f);
  const f4 g  = ((const f4*)gamma)[threadIdx.x];
  const f4 be = ((const f4*)beta)[threadIdx.x];
  f4 o;
#pragma unroll
  for (int j = 0; j < 4; ++j) o[j] = (v[j] - mu) * inv * g[j] + be[j];
  // final output, write-once -> non-temporal
  __builtin_nontemporal_store(o, &((f4*)(out + (size_t)row * D_))[threadIdx.x]);
}

// ---------------------------------------------------------------------------
extern "C" void kernel_launch(void* const* d_in, const int* in_sizes, int n_in,
                              void* d_out, int out_size, void* d_ws, size_t ws_size,
                              hipStream_t stream) {
  (void)in_sizes; (void)n_in; (void)out_size; (void)ws_size;

  const float* q     = (const float*)d_in[0];
  const float* k     = (const float*)d_in[1];
  const float* v     = (const float*)d_in[2];
  const int*   mask  = (const int*)d_in[3];
  const float* Wq    = (const float*)d_in[4];
  const float* bq    = (const float*)d_in[5];
  const float* Wk    = (const float*)d_in[6];
  const float* bk    = (const float*)d_in[7];
  const float* Wv    = (const float*)d_in[8];
  const float* bv    = (const float*)d_in[9];
  const float* Wo    = (const float*)d_in[10];
  const float* bo    = (const float*)d_in[11];
  const float* gamma = (const float*)d_in[12];
  const float* beta  = (const float*)d_in[13];

  const size_t PROJ_ELTS = (size_t)B_ * H_ * S_ * HD_;   // 4,194,304
  const size_t W_ELTS    = (size_t)D_ * D_;              // 1,048,576
  _Float16* qh  = (_Float16*)d_ws;
  _Float16* kh  = qh + PROJ_ELTS;
  _Float16* vT  = kh + PROJ_ELTS;
  _Float16* ctx = vT + PROJ_ELTS;
  float*    xbf = (float*)(ctx + PROJ_ELTS);
  _Float16* Wqh = (_Float16*)(xbf + (size_t)B_ * S_ * D_);
  _Float16* Wkh = Wqh + W_ELTS;
  _Float16* Wvh = Wkh + W_ELTS;
  _Float16* Woh = Wvh + W_ELTS;                          // 56MB total scratch

  float* outLN  = (float*)d_out;                          // [B,S,D]
  float* attOut = outLN + (size_t)B_ * S_ * D_;           // [B,H,S,S]

  // weights -> f16 once (L2-resident afterwards)
  const int n4 = (int)(W_ELTS / 4);
  cvt_f32_f16<<<n4 / 256, 256, 0, stream>>>(Wq, Wqh, n4);
  cvt_f32_f16<<<n4 / 256, 256, 0, stream>>>(Wk, Wkh, n4);
  cvt_f32_f16<<<n4 / 256, 256, 0, stream>>>(Wv, Wvh, n4);
  cvt_f32_f16<<<n4 / 256, 256, 0, stream>>>(Wo, Woh, n4);

  const dim3 gemmGrid(D_ / 64, (B_ * S_) / 32);
  proj_gemm<<<gemmGrid, 64, 0, stream>>>(q, Wqh, bq, qh, 0, 0.125f);
  proj_gemm<<<gemmGrid, 64, 0, stream>>>(k, Wkh, bk, kh, 0, 1.0f);
  proj_gemm<<<gemmGrid, 64, 0, stream>>>(v, Wvh, bv, vT, 1, 1.0f);

  const size_t ldsBytes = (size_t)2 * 16 * S_ * sizeof(float);  // 256KB
  attn_kernel<<<dim3(S_ / 32, B_ * H_), 64, ldsBytes, stream>>>(qh, kh, vT, mask,
                                                                attOut, ctx);

  oproj_gemm<<<gemmGrid, 64, 0, stream>>>(ctx, Woh, bo, q, xbf);
  layernorm_kernel<<<B_ * S_, 256, 0, stream>>>(xbf, gamma, beta, outLN);
}
